// HARHN_9543417332548
// MI455X (gfx1250) — compile-verified, matrix operand order checked by reading, not run on hardware
//
#include <hip/hip_runtime.h>

// ---------------- problem constants ----------------
#define B_   256
#define T_   256
#define F_   16
#define NF_  64
#define EI_  64
#define NE_  128
#define ND_  128
#define RD_  3
#define BT_  (B_*T_)   // 65536

typedef __bf16 bf16_t;
typedef __attribute__((ext_vector_type(8)))  bf16_t bf16x8;
typedef __attribute__((ext_vector_type(16))) bf16_t bf16x16;
typedef __attribute__((ext_vector_type(8)))  float  f32x8;

// ---------------- WMMA helpers ----------------
// 16-bit A (16x32) / B(as N-major rows of K) fragment: per lane two contiguous
// 8-element bf16 runs:  k = kt*32 + half*8 + {0..7}  and  +16.
__device__ __forceinline__ bf16x16 frag_row(const bf16_t* rowp, int kt, int kh) {
  const bf16_t* p = rowp + kt * 32 + kh * 8;
  bf16x8 lo = *(const bf16x8*)(p);
  bf16x8 hi = *(const bf16x8*)(p + 16);
  return __builtin_shufflevector(lo, hi, 0,1,2,3,4,5,6,7,8,9,10,11,12,13,14,15);
}

__device__ __forceinline__ f32x8 wmma_bf16(bf16x16 a, bf16x16 b, f32x8 c) {
  return __builtin_amdgcn_wmma_f32_16x16x32_bf16(false, a, false, b, (short)0, c,
                                                 false, false);
}

__device__ __forceinline__ f32x8 splat8(float v) {
  f32x8 r;
#pragma unroll
  for (int i = 0; i < 8; ++i) r[i] = v;
  return r;
}

__device__ __forceinline__ float sigm(float x) { return 1.f / (1.f + __expf(-x)); }

// Pointer launder: makes the address loop-variant to LICM so loop-invariant
// weight fragments are re-loaded from L2 each iteration instead of being
// hoisted en masse and spilled to scratch.
__device__ __forceinline__ const bf16_t* opaque(const bf16_t* p) {
#if defined(__gfx1250__)
  asm volatile("" : "+v"(p));
#endif
  return p;
}

// CDNA5 async copy global -> LDS (16B per lane). Low 32 bits of a generic
// pointer to LDS are the LDS byte address (flat aperture truncation).
__device__ __forceinline__ void cp_async16(void* lds_dst, const void* gsrc) {
#if defined(__gfx1250__)
  unsigned ldsoff = (unsigned)(uintptr_t)lds_dst;
  asm volatile("global_load_async_to_lds_b128 %0, %1, off"
               :: "v"(ldsoff), "v"(gsrc) : "memory");
#else
  *(bf16x8*)lds_dst = *(const bf16x8*)gsrc;
#endif
}

__device__ __forceinline__ void async_wait0() {
#if defined(__gfx1250__)
  asm volatile("s_wait_asynccnt 0x0" ::: "memory");
#endif
}

// ---------------- conv1d (same pad, k=5) + ReLU ----------------
__global__ void k_conv(const float* __restrict__ xin, const float* __restrict__ w,
                       const float* __restrict__ bias, float* __restrict__ out, int CIN) {
  extern __shared__ char smem[];
  float* sw = (float*)smem;
  const int nw = NF_ * CIN * 5;
  for (int i = threadIdx.x; i < nw; i += 256) sw[i] = w[i];
  __syncthreads();
  long idx = (long)blockIdx.x * 256 + threadIdx.x;   // [b][t][o]
  int o = (int)(idx & 63);
  int t = (int)((idx >> 6) & 255);
  int b = (int)(idx >> 14);
  float acc = bias[o];
  const float* xrow = xin + (long)b * T_ * CIN;
  for (int c = 0; c < CIN; ++c) {
    const float* wr = sw + (o * CIN + c) * 5;
#pragma unroll
    for (int f = 0; f < 5; ++f) {
      int tt = t + f - 2;
      if ((unsigned)tt < (unsigned)T_) acc += xrow[(long)tt * CIN + c] * wr[f];
    }
  }
  out[idx] = fmaxf(acc, 0.f);
}

// ---------------- c2e projection, output bf16 ----------------
__global__ void k_c2e(const float* __restrict__ a, const float* __restrict__ w,
                      const float* __restrict__ bias, bf16_t* __restrict__ xe) {
  long idx = (long)blockIdx.x * 256 + threadIdx.x;   // [bt][e]
  int e = (int)(idx & 63);
  long bt = idx >> 6;
  float acc = bias[e];
  const float* ar = a + bt * NF_;
  for (int c = 0; c < NF_; ++c) acc += ar[c] * w[c * EI_ + e];
  xe[idx] = (bf16_t)acc;
}

// ---------------- weight transpose f32[K][N] -> bf16[N][K] ----------------
__global__ void k_tr(const float* __restrict__ src, bf16_t* __restrict__ dst,
                     int K, int N) {
  long base = (long)blockIdx.y * K * N;
  int i = blockIdx.x * 256 + threadIdx.x;
  if (i < K * N) {
    int nn = i / K, kk = i - nn * K;
    dst[base + (long)nn * K + kk] = (bf16_t)src[base + (long)kk * N + nn];
  }
}

// ---------------- encoder RHN scan (16 batch rows / block, 8 waves) ----------
// All weights live in LDS (229 KB of the 320 KB/WGP), async-staged at entry.
__global__ void __launch_bounds__(256)
k_encoder(const bf16_t* __restrict__ xe,
          const bf16_t* __restrict__ WHt, const bf16_t* __restrict__ WTt,
          const bf16_t* __restrict__ RHwT, const bf16_t* __restrict__ RTwT,
          const float* __restrict__ RHb, const float* __restrict__ RTb,
          bf16_t* __restrict__ hA) {
  extern __shared__ char smem[];
  bf16_t* sWH  = (bf16_t*)smem;                 // 128*64
  bf16_t* sWT  = sWH + NE_ * EI_;               // 128*64
  bf16_t* sRH  = sWT + NE_ * EI_;               // 3*128*128
  bf16_t* sRT  = sRH + RD_ * NE_ * NE_;         // 3*128*128
  bf16_t* sh_s = sRT + RD_ * NE_ * NE_;         // 16*128

  const int tid = threadIdx.x, lane = tid & 31, wv = tid >> 5;
  const int kh = lane >> 4, p = lane & 15;
  const int b0 = blockIdx.x * 16;
  const int n = wv * 16 + p;                 // this lane's output column

  // async-stage all weights into LDS (16B chunks per lane)
  for (int i = tid; i < NE_ * EI_ / 8; i += 256) {
    cp_async16(sWH + i * 8, WHt + i * 8);
    cp_async16(sWT + i * 8, WTt + i * 8);
  }
  for (int i = tid; i < RD_ * NE_ * NE_ / 8; i += 256) {
    cp_async16(sRH + i * 8, RHwT + i * 8);
    cp_async16(sRT + i * 8, RTwT + i * 8);
  }
  for (int i = tid; i < 16 * NE_; i += 256) sh_s[i] = (bf16_t)0.f;
  f32x8 sreg = splat8(0.f);
  async_wait0();
  __syncthreads();

  for (int t = 0; t < T_; ++t) {
#pragma unroll
    for (int l = 0; l < RD_; ++l) {
      f32x8 aH = splat8(RHb[l * NE_ + n]);
      f32x8 aT = splat8(RTb[l * NE_ + n]);
      if (l == 0) {
        const bf16_t* xr = xe + ((long)(b0 + p) * T_ + t) * EI_;
        const bf16_t* bh = sWH + (long)n * EI_;
        const bf16_t* bt = sWT + (long)n * EI_;
#pragma unroll
        for (int kt = 0; kt < 2; ++kt) {
          bf16x16 a = frag_row(xr, kt, kh);
          aH = wmma_bf16(a, frag_row(bh, kt, kh), aH);
          aT = wmma_bf16(a, frag_row(bt, kt, kh), aT);
        }
      }
      const bf16_t* sr = sh_s + p * NE_;
      const bf16_t* rh = sRH + ((long)l * NE_ + n) * NE_;
      const bf16_t* rt = sRT + ((long)l * NE_ + n) * NE_;
#pragma unroll
      for (int kt = 0; kt < 4; ++kt) {
        bf16x16 a = frag_row(sr, kt, kh);
        aH = wmma_bf16(a, frag_row(rh, kt, kh), aH);
        aT = wmma_bf16(a, frag_row(rt, kt, kh), aT);
      }
      __syncthreads();   // all reads of sh_s done
#pragma unroll
      for (int i = 0; i < 8; ++i) {
        float hh = tanhf(aH[i]);
        float tg = sigm(aT[i]);
        float sn = hh * tg + (1.f - tg) * sreg[i];
        sreg[i] = sn;
        int m = i + 8 * kh;
        bf16_t sb = (bf16_t)sn;
        sh_s[m * NE_ + n] = sb;
        hA[((long)l * BT_ + (long)(b0 + m) * T_ + t) * NE_ + n] = sb;
      }
      __syncthreads();
    }
  }
}

// ---------------- Uh = h @ Uk_w + Uk_b  (per r), output bf16 -----------------
__global__ void __launch_bounds__(256)
k_uh(const bf16_t* __restrict__ hA, const bf16_t* __restrict__ UkT,
     const float* __restrict__ Ukb, bf16_t* __restrict__ Uh) {
  const int tid = threadIdx.x, lane = tid & 31, wv = tid >> 5;
  const int kh = lane >> 4, p = lane & 15;
  const int r = blockIdx.y;
  const long row0 = (long)blockIdx.x * 16;
  const int col = wv * 16 + p;
  f32x8 acc = splat8(Ukb[r * NE_ + col]);
  const bf16_t* Ar = hA + ((long)r * BT_ + row0 + p) * NE_;
  const bf16_t* Br = UkT + ((long)r * NE_ + col) * NE_;
#pragma unroll
  for (int kt = 0; kt < 4; ++kt)
    acc = wmma_bf16(frag_row(Ar, kt, kh), frag_row(Br, kt, kh), acc);
#pragma unroll
  for (int i = 0; i < 8; ++i)
    Uh[((long)r * BT_ + row0 + i + 8 * kh) * NE_ + col] = (bf16_t)acc[i];
}

// ---------------- attentive decoder scan -------------------------------------
__global__ void __launch_bounds__(256)
k_decoder(const float* __restrict__ y,
          const bf16_t* __restrict__ hA, const bf16_t* __restrict__ Uh,
          const bf16_t* __restrict__ TkT,
          const bf16_t* __restrict__ RHwT, const bf16_t* __restrict__ RTwT,
          const float* __restrict__ RHb, const float* __restrict__ RTb,
          const float* __restrict__ dWH, const float* __restrict__ dWT,
          const float* __restrict__ vkw, const float* __restrict__ vkb,
          const float* __restrict__ Wtw, const float* __restrict__ Vtw,
          const float* __restrict__ Vtb,
          float* __restrict__ s_out, float* __restrict__ d_out) {
  extern __shared__ char smem[];
  bf16_t* sh_s  = (bf16_t*)smem;               // 16*128 bf16      (4 KB)
  float*  shTsD = (float*)(smem + 4096);       // 3*16*128 f32: Ts then d (24 KB)
  float*  shE   = (float*)(smem + 28672);      // 3*16*256 f32: e then alpha (48 KB)
  float*  shY   = (float*)(smem + 77824);      // 16 f32
  float*  shvk  = (float*)(smem + 77888);      // 3*128 f32

  const int tid = threadIdx.x, lane = tid & 31, wv = tid >> 5;
  const int kh = lane >> 4, p = lane & 15;
  const int b0 = blockIdx.x * 16;
  const int n = wv * 16 + p;
  const float wt_s = Wtw[0];
  const float vtb  = Vtb[0];

  for (int i = tid; i < RD_ * NE_; i += 256) shvk[i] = vkw[i];
  for (int i = tid; i < 16 * ND_; i += 256) sh_s[i] = (bf16_t)0.f;
  f32x8 sreg = splat8(0.f);
  __syncthreads();

  for (int t = 0; t < T_; ++t) {
    // 1. Ts[r] = s @ Tk_w[r]     (24 wave-tile jobs)
    for (int job = wv; job < 24; job += 8) {
      int r = job >> 3, nt = job & 7;
      int col = nt * 16 + p;
      f32x8 acc = splat8(0.f);
      const bf16_t* Ar = sh_s + p * ND_;
      const bf16_t* Br = opaque(TkT + ((long)r * NE_ + col) * ND_);
#pragma unroll
      for (int kt = 0; kt < 4; ++kt)
        acc = wmma_bf16(frag_row(Ar, kt, kh), frag_row(Br, kt, kh), acc);
#pragma unroll
      for (int i = 0; i < 8; ++i)
        shTsD[(r * 16 + i + 8 * kh) * NE_ + col] = acc[i];
    }
    __syncthreads();

    // 2. e[r,b,t'] = vk . tanh(Uh + Ts) + vk_b   (the hot loop)
    for (int idx = tid; idx < RD_ * 16 * T_; idx += 256) {
      int tp = idx & (T_ - 1);
      int b  = (idx >> 8) & 15;
      int r  = idx >> 12;
      const bf16_t* uh = Uh + ((long)r * BT_ + (long)(b0 + b) * T_ + tp) * NE_;
      if (b < 15) __builtin_prefetch(uh + (long)T_ * NE_, 0, 1);
      const float* ts  = shTsD + (r * 16 + b) * NE_;
      const float* vkr = shvk + r * NE_;
      float acc = 0.f;
#pragma unroll 2
      for (int m = 0; m < NE_; m += 8) {
        bf16x8 u = *(const bf16x8*)(uh + m);
#pragma unroll
        for (int q = 0; q < 8; ++q)
          acc += vkr[m + q] * tanhf((float)u[q] + ts[m + q]);
      }
      shE[idx] = acc + vkb[r];
    }
    __syncthreads();

    // 3. softmax over t' per (r,b)
    if (tid < RD_ * 16) {
      float* e = shE + tid * T_;
      float mx = -1e30f;
      for (int tp = 0; tp < T_; ++tp) mx = fmaxf(mx, e[tp]);
      float sm = 0.f;
      for (int tp = 0; tp < T_; ++tp) { float v = __expf(e[tp] - mx); e[tp] = v; sm += v; }
      float inv = 1.f / sm;
      for (int tp = 0; tp < T_; ++tp) e[tp] *= inv;
    }
    __syncthreads();

    // 4. d[r,b,n] = sum_t' alpha * h   (overwrites Ts region)
    for (int idx = tid; idx < RD_ * 16 * NE_; idx += 256) {
      int nn = idx & (NE_ - 1);
      int b  = (idx >> 7) & 15;
      int r  = idx >> 11;
      const bf16_t* hr = hA + ((long)r * BT_ + (long)(b0 + b) * T_) * NE_ + nn;
      const float* al = shE + (r * 16 + b) * T_;
      float acc = 0.f;
      for (int tp = 0; tp < T_; ++tp) acc += al[tp] * (float)hr[(long)tp * NE_];
      shTsD[(r * 16 + b) * NE_ + nn] = acc;
    }
    __syncthreads();

    // 5. y_til[b] (TF==1 => scalar per batch)
    if (tid < 16) {
      int b = tid;
      float acc = y[(long)(b0 + b) * T_ + t] * wt_s + vtb;
      for (int r = 0; r < RD_; ++r)
        for (int nn = 0; nn < NE_; ++nn)
          acc += shTsD[(r * 16 + b) * NE_ + nn] * Vtw[r * NE_ + nn];
      shY[b] = acc;
    }
    __syncthreads();

    // 6. decoder RHN cell (x-input is the scalar y_til)
#pragma unroll
    for (int l = 0; l < RD_; ++l) {
      f32x8 aH = splat8(RHb[l * ND_ + n]);
      f32x8 aT = splat8(RTb[l * ND_ + n]);
      if (l == 0) {
        float whn = dWH[n], wtn = dWT[n];
#pragma unroll
        for (int i = 0; i < 8; ++i) {
          float yv = shY[i + 8 * kh];
          aH[i] += yv * whn;
          aT[i] += yv * wtn;
        }
      }
      const bf16_t* sr = sh_s + p * ND_;
      const bf16_t* rh = opaque(RHwT + ((long)l * ND_ + n) * ND_);
      const bf16_t* rt = opaque(RTwT + ((long)l * ND_ + n) * ND_);
#pragma unroll
      for (int kt = 0; kt < 4; ++kt) {
        bf16x16 a = frag_row(sr, kt, kh);
        aH = wmma_bf16(a, frag_row(rh, kt, kh), aH);
        aT = wmma_bf16(a, frag_row(rt, kt, kh), aT);
      }
      __syncthreads();
#pragma unroll
      for (int i = 0; i < 8; ++i) {
        float hh = tanhf(aH[i]);
        float tg = sigm(aT[i]);
        float sn = hh * tg + (1.f - tg) * sreg[i];
        sreg[i] = sn;
        sh_s[(i + 8 * kh) * ND_ + n] = (bf16_t)sn;
      }
      __syncthreads();
    }
  }

  // final s and d
#pragma unroll
  for (int i = 0; i < 8; ++i)
    s_out[(long)(b0 + i + 8 * kh) * ND_ + n] = sreg[i];
  for (int idx = tid; idx < RD_ * 16 * NE_; idx += 256) {
    int nn = idx & (NE_ - 1);
    int b  = (idx >> 7) & 15;
    int r  = idx >> 11;
    d_out[(long)(b0 + b) * (RD_ * NE_) + r * NE_ + nn] = shTsD[(r * 16 + b) * NE_ + nn];
  }
}

// ---------------- final projection -------------------------------------------
__global__ void k_final(const float* __restrict__ s, const float* __restrict__ d,
                        const float* __restrict__ Ww, const float* __restrict__ Wb,
                        const float* __restrict__ Vw, const float* __restrict__ Vb,
                        float* __restrict__ out) {
  int b = threadIdx.x;
  if (b < B_) {
    float acc = Wb[0] + Vb[0];
    for (int nn = 0; nn < ND_; ++nn) acc += s[(long)b * ND_ + nn] * Ww[nn];
    for (int j = 0; j < RD_ * NE_; ++j) acc += d[(long)b * RD_ * NE_ + j] * Vw[j];
    out[b] = acc;
  }
}

// ---------------- host launcher ----------------------------------------------
extern "C" void kernel_launch(void* const* d_in, const int* in_sizes, int n_in,
                              void* d_out, int out_size, void* d_ws, size_t ws_size,
                              hipStream_t stream) {
  (void)in_sizes; (void)n_in; (void)out_size; (void)ws_size;
  const float* x      = (const float*)d_in[0];
  const float* y      = (const float*)d_in[1];
  const float* conv0w = (const float*)d_in[2];
  const float* conv0b = (const float*)d_in[3];
  const float* conv1w = (const float*)d_in[4];
  const float* conv1b = (const float*)d_in[5];
  const float* c2ew   = (const float*)d_in[6];
  const float* c2eb   = (const float*)d_in[7];
  const float* encWH  = (const float*)d_in[8];
  const float* encWT  = (const float*)d_in[9];
  const float* encRHw = (const float*)d_in[10];
  const float* encRHb = (const float*)d_in[11];
  const float* encRTw = (const float*)d_in[12];
  const float* encRTb = (const float*)d_in[13];
  const float* decWH  = (const float*)d_in[14];
  const float* decWT  = (const float*)d_in[15];
  const float* decRHw = (const float*)d_in[16];
  const float* decRHb = (const float*)d_in[17];
  const float* decRTw = (const float*)d_in[18];
  const float* decRTb = (const float*)d_in[19];
  const float* Tkw    = (const float*)d_in[20];
  const float* Ukw    = (const float*)d_in[21];
  const float* Ukb    = (const float*)d_in[22];
  const float* vkw    = (const float*)d_in[23];
  const float* vkb    = (const float*)d_in[24];
  const float* Wtw    = (const float*)d_in[25];
  const float* Vtw    = (const float*)d_in[26];
  const float* Vtb    = (const float*)d_in[27];
  const float* Ww     = (const float*)d_in[28];
  const float* Wb     = (const float*)d_in[29];
  const float* Vw     = (const float*)d_in[30];
  const float* Vb     = (const float*)d_in[31];

  char* ws = (char*)d_ws;
  size_t off = 0;
  auto carve = [&](size_t bytes) -> char* {
    char* pp = ws + off;
    off = (off + bytes + 255) & ~(size_t)255;
    return pp;
  };
  float*  act0 = (float*)carve((size_t)BT_ * NF_ * 4);
  float*  act1 = (float*)carve((size_t)BT_ * NF_ * 4);
  bf16_t* xe   = (bf16_t*)carve((size_t)BT_ * EI_ * 2);
  bf16_t* WHt  = (bf16_t*)carve((size_t)EI_ * NE_ * 2);
  bf16_t* WTt  = (bf16_t*)carve((size_t)EI_ * NE_ * 2);
  bf16_t* eRH  = (bf16_t*)carve((size_t)RD_ * NE_ * NE_ * 2);
  bf16_t* eRT  = (bf16_t*)carve((size_t)RD_ * NE_ * NE_ * 2);
  bf16_t* TkT  = (bf16_t*)carve((size_t)RD_ * NE_ * ND_ * 2);
  bf16_t* dRH  = (bf16_t*)carve((size_t)RD_ * ND_ * ND_ * 2);
  bf16_t* dRT  = (bf16_t*)carve((size_t)RD_ * ND_ * ND_ * 2);
  bf16_t* UkT  = (bf16_t*)carve((size_t)RD_ * NE_ * NE_ * 2);
  bf16_t* hA   = (bf16_t*)carve((size_t)RD_ * BT_ * NE_ * 2);
  bf16_t* Uh   = (bf16_t*)carve((size_t)RD_ * BT_ * NE_ * 2);
  float*  sOut = (float*)carve((size_t)B_ * ND_ * 4);
  float*  dOut = (float*)carve((size_t)B_ * RD_ * NE_ * 4);

  // front-end
  k_conv<<<BT_ * NF_ / 256, 256, NF_ * F_ * 5 * 4, stream>>>(x, conv0w, conv0b, act0, F_);
  k_conv<<<BT_ * NF_ / 256, 256, NF_ * NF_ * 5 * 4, stream>>>(act0, conv1w, conv1b, act1, NF_);
  k_c2e<<<BT_ * EI_ / 256, 256, 0, stream>>>(act1, c2ew, c2eb, xe);

  // weight packing (bf16, transposed [N][K])
  k_tr<<<dim3((EI_ * NE_ + 255) / 256, 1), 256, 0, stream>>>(encWH, WHt, EI_, NE_);
  k_tr<<<dim3((EI_ * NE_ + 255) / 256, 1), 256, 0, stream>>>(encWT, WTt, EI_, NE_);
  k_tr<<<dim3((NE_ * NE_ + 255) / 256, RD_), 256, 0, stream>>>(encRHw, eRH, NE_, NE_);
  k_tr<<<dim3((NE_ * NE_ + 255) / 256, RD_), 256, 0, stream>>>(encRTw, eRT, NE_, NE_);
  k_tr<<<dim3((ND_ * NE_ + 255) / 256, RD_), 256, 0, stream>>>(Tkw, TkT, ND_, NE_);
  k_tr<<<dim3((ND_ * ND_ + 255) / 256, RD_), 256, 0, stream>>>(decRHw, dRH, ND_, ND_);
  k_tr<<<dim3((ND_ * ND_ + 255) / 256, RD_), 256, 0, stream>>>(decRTw, dRT, ND_, ND_);
  k_tr<<<dim3((NE_ * NE_ + 255) / 256, RD_), 256, 0, stream>>>(Ukw, UkT, NE_, NE_);

  // encoder scan (weights async-staged into 229 KB of LDS) + attention precompute
  size_t encSm = (size_t)(2 * NE_ * EI_ + 2 * RD_ * NE_ * NE_ + 16 * NE_) * 2; // 233472
  k_encoder<<<B_ / 16, 256, encSm, stream>>>(xe, WHt, WTt, eRH, eRT, encRHb, encRTb, hA);
  k_uh<<<dim3(BT_ / 16, RD_), 256, 0, stream>>>(hA, UkT, Ukb, Uh);

  // decoder scan
  size_t decSm = 77888 + (size_t)RD_ * NE_ * 4;   // 79424 bytes
  k_decoder<<<B_ / 16, 256, decSm, stream>>>(y, hA, Uh, TkT, dRH, dRT,
                                             decRHb, decRTb, decWH, decWT,
                                             vkw, vkb, Wtw, Vtw, Vtb, sOut, dOut);

  // output head
  k_final<<<1, 256, 0, stream>>>(sOut, dOut, Ww, Wb, Vw, Vb, (float*)d_out);
}
